// Encoder_35064113004946
// MI455X (gfx1250) — matrix operand order repaired and checked by reading, not compiled, not run on hardware
//
#include <hip/hip_runtime.h>
#include <hip/hip_bf16.h>

// ---------------- types ----------------
typedef __attribute__((ext_vector_type(16))) __bf16    v16bf;
typedef __attribute__((ext_vector_type(8)))  float     v8f;
typedef __attribute__((ext_vector_type(8)))  unsigned  v8u;

// ---------------- bf16 helpers ----------------
static __device__ __forceinline__ unsigned f2bf1(float a) {
    unsigned u = __builtin_bit_cast(unsigned, a);
    return (u + 0x7FFFu + ((u >> 16) & 1u)) >> 16;
}
static __device__ __forceinline__ unsigned f2bf_pk(float a, float b) {
    return (f2bf1(a) & 0xFFFFu) | (f2bf1(b) << 16);
}
static __device__ __forceinline__ float bf2f(unsigned u16) {
    return __builtin_bit_cast(float, u16 << 16);
}

// ---------------- WMMA fragment loads ----------------
// A (16x32 bf16) from LDS: lane l -> row m=l&15, group g=l>>4.
// dwords: rowbase + kt*16 + g*4 + {0..3} and + 8 + {0..3}
static __device__ __forceinline__ v16bf load_a_lds(const unsigned* rowbase, int o) {
    uint4 lo = *(const uint4*)(rowbase + o);
    uint4 hi = *(const uint4*)(rowbase + o + 8);
    v8u u = {lo.x, lo.y, lo.z, lo.w, hi.x, hi.y, hi.z, hi.w};
    return __builtin_bit_cast(v16bf, u);
}
// B (32x16 bf16) from pre-swizzled global fragment buffer: 8 dwords/lane contiguous
static __device__ __forceinline__ v16bf load_b_gbl(const unsigned* p) {
    uint4 lo = *(const uint4*)(p);
    uint4 hi = *(const uint4*)(p + 4);
    v8u u = {lo.x, lo.y, lo.z, lo.w, hi.x, hi.y, hi.z, hi.w};
    return __builtin_bit_cast(v16bf, u);
}
static __device__ __forceinline__ v8f wmma_bf16(v16bf a, v16bf b, v8f c) {
    return __builtin_amdgcn_wmma_f32_16x16x32_bf16(false, a, false, b, (short)0, c, false, false);
}

// ---------------- prep kernels ----------------
// x0[u=bn*128+t][f] (bf16, packed dwords) = data[t][bn][f]   (data: [128,256,128] f32)
__global__ void prep_x0_kernel(const float* __restrict__ data, unsigned* __restrict__ x0) {
    int i = blockIdx.x * blockDim.x + threadIdx.x;          // 32768*64 dwords
    if (i >= 32768 * 64) return;
    int u = i >> 6, d = i & 63;
    int bn = u >> 7, t = u & 127;
    const float* p = data + ((size_t)(t * 256 + bn)) * 128 + 2 * d;
    x0[i] = f2bf_pk(p[0], p[1]);
}

// W [K2 x 256] f32 -> fragment layout [kt][nt][lane][v] (dword = bf16 pair K,K+1)
template <int KT>
__global__ void prep_w_kernel(const float* __restrict__ W, unsigned* __restrict__ wf) {
    int tid = blockIdx.x * blockDim.x + threadIdx.x;
    if (tid >= KT * 16 * 32 * 8) return;
    int v = tid & 7, lane = (tid >> 3) & 31, nt = (tid >> 8) & 15, kt = tid >> 12;
    int n = lane & 15, g = lane >> 4;
    int k = kt * 32 + g * 8 + 2 * (v & 3) + 16 * (v >> 2);
    int col = nt * 16 + n;
    wf[tid] = f2bf_pk(W[(size_t)k * 256 + col], W[(size_t)(k + 1) * 256 + col]);
}

// Wc [o=256][i=256][tap=5] f32 -> [tap][kt(8)][nt(16)][lane][v]  (K dim = i, N dim = o)
__global__ void prep_wc_kernel(const float* __restrict__ Wc, unsigned* __restrict__ wf) {
    int tid = blockIdx.x * blockDim.x + threadIdx.x;        // 5*8*16*32*8 = 163840
    if (tid >= 163840) return;
    int v = tid & 7, lane = (tid >> 3) & 31, nt = (tid >> 8) & 15;
    int kt = (tid >> 12) & 7, tap = tid >> 15;
    int n = lane & 15, g = lane >> 4;
    int i = kt * 32 + g * 8 + 2 * (v & 3) + 16 * (v >> 2);
    int o = nt * 16 + n;
    wf[tid] = f2bf_pk(Wc[((size_t)o * 256 + i) * 5 + tap],
                      Wc[((size_t)o * 256 + i + 1) * 5 + tap]);
}

__global__ void zero_kernel(float* __restrict__ p, int n) {
    for (int i = blockIdx.x * blockDim.x + threadIdx.x; i < n; i += gridDim.x * blockDim.x)
        p[i] = 0.0f;
}

// ---------------- EdgeConv ----------------
// 64 edges/block (4 M-tiles), 128 threads (4 waves x 4 n-tiles). One B fragment feeds
// 4 WMMAs (one per M-tile) -> 4x less B-fragment L2 traffic per FLOP.
// __launch_bounds__(128,1): lift VGPR cap so 16 v8f accumulators live in registers.
// out[dst] = segment-max(relu(A@W + b)) via integer atomicMax (values >= 0, out pre-zeroed).
template <int C>
__global__ void __launch_bounds__(128, 1)
edgeconv_kernel(const unsigned* __restrict__ x,     // [nodes][C/2] dwords bf16
                const int* __restrict__ src,
                const int* __restrict__ dst,
                const unsigned* __restrict__ wfrag, // [KT][16][32][8]
                const float* __restrict__ bias,     // [256]
                float* __restrict__ out)            // [nodes][256]
{
    constexpr int K2 = 2 * C, KT = K2 / 32, CD = C / 2, AS = K2 / 2 + 4;
    __shared__ __align__(16) unsigned sA[64 * AS];
    __shared__ int sS[64];
    __shared__ int sD[64];

    const int tid = threadIdx.x;
    if (tid < 64) {
        int e = blockIdx.x * 64 + tid;
        sS[tid] = src[e];
        sD[tid] = dst[e];
    }
    __syncthreads();

    // stage A: row r = [ x_i (CD dwords) | x_j - x_i (CD dwords) ]
    for (int i = tid; i < 64 * CD; i += 128) {
        int r = i / CD, d = i % CD;
        unsigned wi = x[(size_t)(unsigned)sD[r] * CD + d];
        unsigned wj = x[(size_t)(unsigned)sS[r] * CD + d];
        sA[r * AS + d] = wi;
        float dl = bf2f(wj & 0xFFFFu) - bf2f(wi & 0xFFFFu);
        float dh = bf2f(wj >> 16)     - bf2f(wi >> 16);
        sA[r * AS + CD + d] = f2bf_pk(dl, dh);
    }
    __syncthreads();

    const int wave = tid >> 5, lane = tid & 31;
    const int n = lane & 15, g = lane >> 4;

    v8f acc[16];                                  // [mt][j]
    const v8f vzero = {0.f, 0.f, 0.f, 0.f, 0.f, 0.f, 0.f, 0.f};
#pragma unroll
    for (int q = 0; q < 16; ++q) acc[q] = vzero;

    for (int kt = 0; kt < KT; ++kt) {
        v16bf a[4];
#pragma unroll
        for (int mt = 0; mt < 4; ++mt)
            a[mt] = load_a_lds(sA + (mt * 16 + n) * AS, kt * 16 + g * 4);
        if (kt + 1 < KT)  // warm caches for next K-step's B fragments
            __builtin_prefetch(wfrag + (((size_t)(kt + 1) * 16 + wave * 4) * 32 + lane) * 8, 0, 0);
#pragma unroll
        for (int j = 0; j < 4; ++j) {
            int nt = wave * 4 + j;
            v16bf b = load_b_gbl(wfrag + (((size_t)kt * 16 + nt) * 32 + lane) * 8);
#pragma unroll
            for (int mt = 0; mt < 4; ++mt)
                acc[mt * 4 + j] = wmma_bf16(a[mt], b, acc[mt * 4 + j]);
        }
    }

#pragma unroll
    for (int j = 0; j < 4; ++j) {
        int col = (wave * 4 + j) * 16 + n;
        float bv = bias[col];
#pragma unroll
        for (int mt = 0; mt < 4; ++mt) {
#pragma unroll
            for (int v = 0; v < 8; ++v) {
                float val = fmaxf(acc[mt * 4 + j][v] + bv, 0.0f);   // relu
                int row = sD[mt * 16 + v + 8 * g];                  // D row m = v + 8*g
                atomicMax((int*)&out[(size_t)row * 256 + col], __float_as_int(val));
            }
        }
    }
}

// ---------------- TempConv: Conv1d(k=5,pad=2)+ReLU+MaxPool(2) ----------------
// grid (L/(MT*16), 256bn), 128 threads, MT position-tiles per block.
// One Wc B-fragment feeds MT WMMAs.
template <int L, int MT, bool OUTBF>
__global__ void __launch_bounds__(128, 1)
tempconv_kernel(const float* __restrict__ in,       // [bn*L + t][256]
                const unsigned* __restrict__ wcf,   // [5][8][16][32][8]
                const float* __restrict__ bias,     // [256]
                void* __restrict__ outv)            // bf16 or f32 [bn*(L/2)+p][256]
{
    constexpr int XS = 132;                                  // 128 dwords + pad
    constexpr int R = MT * 16 + 4;                           // staged rows incl. halo
    __shared__ __align__(16) unsigned sX[R * XS];
    const int bn = blockIdx.y, l0 = blockIdx.x * (MT * 16);
    const int tid = threadIdx.x;

    for (int i = tid; i < R * 128; i += 128) {
        int r = i >> 7, d = i & 127;
        int t = l0 - 2 + r;
        unsigned w = 0u;
        if (t >= 0 && t < L) {
            const float* p = in + ((size_t)(bn * L + t)) * 256 + 2 * d;
            w = f2bf_pk(p[0], p[1]);
        }
        sX[r * XS + d] = w;
    }
    __syncthreads();

    const int wave = tid >> 5, lane = tid & 31;
    const int n = lane & 15, g = lane >> 4;

    v8f acc[MT * 4];                                         // [mt][j]
    const v8f vzero = {0.f, 0.f, 0.f, 0.f, 0.f, 0.f, 0.f, 0.f};
#pragma unroll
    for (int q = 0; q < MT * 4; ++q) acc[q] = vzero;

    for (int tap = 0; tap < 5; ++tap) {
        for (int kt = 0; kt < 8; ++kt) {
            v16bf a[MT];
#pragma unroll
            for (int mt = 0; mt < MT; ++mt)                  // row mt*16+m+tap -> pos l0+..+tap-2
                a[mt] = load_a_lds(sX + (mt * 16 + n + tap) * XS, kt * 16 + g * 4);
#pragma unroll
            for (int j = 0; j < 4; ++j) {
                int nt = wave * 4 + j;
                v16bf b = load_b_gbl(wcf + ((((size_t)tap * 8 + kt) * 16 + nt) * 32 + lane) * 8);
#pragma unroll
                for (int mt = 0; mt < MT; ++mt)
                    acc[mt * 4 + j] = wmma_bf16(a[mt], b, acc[mt * 4 + j]);
            }
        }
    }

#pragma unroll
    for (int j = 0; j < 4; ++j) {
        int col = (wave * 4 + j) * 16 + n;
        float bv = bias[col];
#pragma unroll
        for (int mt = 0; mt < MT; ++mt) {
            float y[8];
#pragma unroll
            for (int v = 0; v < 8; ++v) y[v] = fmaxf(acc[mt * 4 + j][v] + bv, 0.0f);
#pragma unroll
            for (int v2 = 0; v2 < 4; ++v2) {                 // pool rows (2v2, 2v2+1)
                float m = fmaxf(y[2 * v2], y[2 * v2 + 1]);
                int prow = bn * (L / 2) + l0 / 2 + mt * 8 + g * 4 + v2;
                if (OUTBF)
                    ((unsigned short*)outv)[(size_t)prow * 256 + col] = (unsigned short)f2bf1(m);
                else
                    ((float*)outv)[(size_t)prow * 256 + col] = m;
            }
        }
    }
}

// ---------------- launch ----------------
extern "C" void kernel_launch(void* const* d_in, const int* in_sizes, int n_in,
                              void* d_out, int out_size, void* d_ws, size_t ws_size,
                              hipStream_t stream) {
    const float* data = (const float*)d_in[0];
    const int*   ei0  = (const int*)d_in[2];     // (2, 262144)
    const int*   ei1  = (const int*)d_in[3];     // (2, 131072)
    const float* W0   = (const float*)d_in[4];
    const float* b0   = (const float*)d_in[5];
    const float* Wc0  = (const float*)d_in[6];
    const float* bc0  = (const float*)d_in[7];
    const float* W1   = (const float*)d_in[8];
    const float* b1   = (const float*)d_in[9];
    const float* Wc1  = (const float*)d_in[10];
    const float* bc1  = (const float*)d_in[11];

    const int E0 = 262144, E1 = 131072;
    const int N0 = 32768, N1 = 16384;

    char* ws = (char*)d_ws;
    size_t off = 0;
    auto carve = [&](size_t bytes) { char* p = ws + off; off += (bytes + 255) & ~(size_t)255; return p; };
    unsigned* x0   = (unsigned*)carve((size_t)N0 * 64 * 4);        // 8 MB   bf16 [32768,128]
    unsigned* wf0  = (unsigned*)carve((size_t)8 * 4096 * 4);       // 128 KB
    unsigned* wcf0 = (unsigned*)carve((size_t)163840 * 4);         // 640 KB
    float*    out0 = (float*)carve((size_t)N0 * 256 * 4);          // 32 MB
    unsigned* x1   = (unsigned*)carve((size_t)N1 * 128 * 4);       // 8 MB   bf16 [16384,256]
    unsigned* wf1  = (unsigned*)carve((size_t)16 * 4096 * 4);      // 256 KB
    unsigned* wcf1 = (unsigned*)carve((size_t)163840 * 4);         // 640 KB
    float*    out1 = (float*)carve((size_t)N1 * 256 * 4);          // 16 MB
    (void)ws_size; (void)in_sizes; (void)n_in; (void)out_size;

    // prep
    prep_x0_kernel<<<(N0 * 64 + 255) / 256, 256, 0, stream>>>(data, x0);
    prep_w_kernel<8><<<(8 * 4096 + 255) / 256, 256, 0, stream>>>(W0, wf0);
    prep_w_kernel<16><<<(16 * 4096 + 255) / 256, 256, 0, stream>>>(W1, wf1);
    prep_wc_kernel<<<(163840 + 255) / 256, 256, 0, stream>>>(Wc0, wcf0);
    prep_wc_kernel<<<(163840 + 255) / 256, 256, 0, stream>>>(Wc1, wcf1);
    zero_kernel<<<2048, 256, 0, stream>>>(out0, N0 * 256);
    zero_kernel<<<2048, 256, 0, stream>>>(out1, N1 * 256);

    // layer 0
    edgeconv_kernel<128><<<E0 / 64, 128, 0, stream>>>(x0, ei0, ei0 + E0, wf0, b0, out0);
    tempconv_kernel<128, 4, true><<<dim3(2, 256), 128, 0, stream>>>(out0, wcf0, bc0, (void*)x1);

    // layer 1
    edgeconv_kernel<256><<<E1 / 64, 128, 0, stream>>>(x1, ei1, ei1 + E1, wf1, b1, out1);
    tempconv_kernel<64, 2, false><<<dim3(2, 256), 128, 0, stream>>>(out1, wcf1, bc1, d_out);
}
